// CubicModel_16587163697191
// MI455X (gfx1250) — matrix operand order repaired and checked by reading, not compiled, not run on hardware
//
#include <hip/hip_runtime.h>
#include <hip/hip_bf16.h>

// ---------------------------------------------------------------------------
// CubicModel on MI455X (gfx1250, wave32, WMMA):
//   h   = cubic_feats(x) @ W0^T + b0     (512x132352 @ 132352x512)
//   out = cubic_feats(h) @ W1^T + b1     (512x132352 @ 132352x256)
// Weights (406 MB f32) are the only HBM traffic (~17.5us floor @ 23.3 TB/s);
// features are generated on the fly into LDS and consumed by
// v_wmma_f32_16x16x32_bf16. Since D and D+NQUAD are multiples of BK, every
// K-tile has a uniform feature type (scalar branch, zero divergence), and the
// triu row index advances incrementally (no sqrt). 32x64 wave tiles give
// 8 WMMAs per 12 ds_load_b128. Split-K(16) + global_atomic_add_f32 provide
// 256/128 blocks to saturate bandwidth.
// ---------------------------------------------------------------------------

typedef __attribute__((ext_vector_type(16))) __bf16 v16bf;
typedef __attribute__((ext_vector_type(8)))  __bf16 v8bf;
typedef __attribute__((ext_vector_type(4)))  __bf16 v4bf;
typedef __attribute__((ext_vector_type(8)))  float  v8f;

constexpr int D      = 512;
constexpr int NQUAD  = (D * D + D) / 2;      // 131328
constexpr int KTOT   = D + NQUAD + D;        // 132352
constexpr int BK     = 64;
constexpr int KTILES = KTOT / BK;            // 2068 (exact)
// tile-type boundaries are BK-aligned: tiles [0,8) linear, [8,2060) quad, [2060,2068) cube
constexpr int TLIN   = D / BK;               // 8
constexpr int TQUAD  = (D + NQUAD) / BK;     // 2060
constexpr int BM     = 64;
constexpr int BN     = 256;
constexpr int SPLITK = 16;
constexpr int LDA    = BK + 8;               // 72 elems = 144B stride -> conflict-free b128 reads

__device__ __forceinline__ __bf16 to_bf16(float f) {
    union { float f; unsigned u; } in; in.f = f;
    unsigned r = in.u + 0x7FFFu + ((in.u >> 16) & 1u);   // round-to-nearest-even
    union { unsigned short s; __bf16 b; } out;
    out.s = (unsigned short)(r >> 16);
    return out.b;
}

__device__ __forceinline__ int triu_rowstart(int i) {
    return i * D - (i * (i - 1)) / 2;  // #pairs before row i of triu(D)
}

__global__ __launch_bounds__(256) void cubic_gemm_wmma(
    const float* __restrict__ X,     // [M][D] feature source (x or h), L2-resident
    const float* __restrict__ W,     // [N][KTOT] row-major weights (streamed)
    const float* __restrict__ bias,  // [N]
    float*       __restrict__ OUT,   // [M][N], pre-zeroed, atomically accumulated
    int N)
{
    __shared__ __align__(32) __bf16 As[BM][LDA];
    __shared__ __align__(32) __bf16 Bs[BN][LDA];

    const int tid   = threadIdx.x;
    const int lane  = tid & 31;
    const int wave  = tid >> 5;
    const int waveM = wave & 1;    // 2 waves along M (32 rows each)
    const int waveN = wave >> 1;   // 4 waves along N (64 cols each)

    const int m0 = blockIdx.x * BM;
    const int n0 = blockIdx.y * BN;

    // split-K tile range for this z-slice
    const int z     = blockIdx.z;
    const int qbase = KTILES / SPLITK;
    const int qrem  = KTILES % SPLITK;
    const int t_begin = z * qbase + (z < qrem ? z : qrem);
    const int t_end   = t_begin + qbase + (z < qrem ? 1 : 0);

    // accumulators: 2x4 tiles of 16x16 f32 (8 VGPRs each); bias folded into z==0
    v8f acc[2][4];
#pragma unroll
    for (int mt = 0; mt < 2; ++mt)
#pragma unroll
        for (int nt = 0; nt < 4; ++nt) {
            float bv = 0.0f;
            if (z == 0) bv = bias[n0 + waveN * 64 + nt * 16 + (lane & 15)];
#pragma unroll
            for (int v = 0; v < 8; ++v) acc[mt][nt][v] = bv;
        }

    const int kkgen = tid & (BK - 1);  // feature column in tile (0..63)
    const int rgen0 = tid >> 6;        // row phase (0..3)
    int iq = 0;                        // incremental triu row (monotonic in p)

    for (int t = t_begin; t < t_end; ++t) {
        const int k0 = t * BK;
        __syncthreads();

        // ---- A tile: cubic features, 64 rows x 64 cols; tile type is uniform ----
        if (t < TLIN) {                       // linear features x_f
            const int ia = k0 + kkgen;
#pragma unroll
            for (int s = 0; s < BM / 4; ++s) {
                const int r = rgen0 + (s << 2);
                As[r][kkgen] = to_bf16(X[(size_t)(m0 + r) * D + ia]);
            }
        } else if (t < TQUAD) {               // quad features x_i * x_j
            const int p = k0 + kkgen - D;
            while (triu_rowstart(iq + 1) <= p) ++iq;     // amortized < 1 iter/tile
            const int ja = iq + (p - triu_rowstart(iq));
#pragma unroll
            for (int s = 0; s < BM / 4; ++s) {
                const int r = rgen0 + (s << 2);
                const float* xr = X + (size_t)(m0 + r) * D;
                As[r][kkgen] = to_bf16(xr[iq] * xr[ja]);
            }
        } else {                              // cube features x_c^3
            const int ia = k0 + kkgen - (D + NQUAD);
#pragma unroll
            for (int s = 0; s < BM / 4; ++s) {
                const int r = rgen0 + (s << 2);
                const float u = X[(size_t)(m0 + r) * D + ia];
                As[r][kkgen] = to_bf16(u * u * u);
            }
        }

        // ---- B tile: stream 256x64 f32 weights -> bf16 LDS (the HBM traffic) ----
        {
            const float* Wk = W + k0;
            for (int idx = tid; idx < BN * (BK / 4); idx += 256) {
                const int row = idx >> 4;
                const int c4  = (idx & 15) << 2;
                const float* src = Wk + (size_t)(n0 + row) * KTOT + c4;
                const float4 w = *(const float4*)src;
                if (t + 1 < t_end) __builtin_prefetch(src + BK, 0, 1);  // global_prefetch_b8
                v4bf b4;
                b4[0] = to_bf16(w.x); b4[1] = to_bf16(w.y);
                b4[2] = to_bf16(w.z); b4[3] = to_bf16(w.w);
                *(v4bf*)&Bs[row][c4] = b4;
            }
        }
        __syncthreads();

        // ---- compute: 2 k-steps of 32; 8 WMMAs per step (wave tile 32x64) ----
        union AB { v16bf v; v8bf h[2]; };
#pragma unroll
        for (int ks = 0; ks < 2; ++ks) {
            AB a[2], b[4];
            // A frag (16x32 bf16): lane = row M, K = (lane/16)*8 + {0..7, 16..23}
            const int ar = waveM * 32 + (lane & 15);
            const int ak = ks * 32 + ((lane >> 4) << 3);
            a[0].h[0] = *(const v8bf*)&As[ar][ak];
            a[0].h[1] = *(const v8bf*)&As[ar][ak + 16];
            a[1].h[0] = *(const v8bf*)&As[ar + 16][ak];
            a[1].h[1] = *(const v8bf*)&As[ar + 16][ak + 16];
            // B frag (32x16 bf16): lane = col N, K = (lane/16)*8 + {0..7, 16..23}
            const int br = waveN * 64 + (lane & 15);
            const int bk = ks * 32 + ((lane >> 4) << 3);
#pragma unroll
            for (int nt = 0; nt < 4; ++nt) {
                b[nt].h[0] = *(const v8bf*)&Bs[br + nt * 16][bk];
                b[nt].h[1] = *(const v8bf*)&Bs[br + nt * 16][bk + 16];
            }
#pragma unroll
            for (int mt = 0; mt < 2; ++mt)
#pragma unroll
                for (int nt = 0; nt < 4; ++nt)
                    acc[mt][nt] = __builtin_amdgcn_wmma_f32_16x16x32_bf16(
                        false, a[mt].v, false, b[nt].v,
                        (short)0, acc[mt][nt], false, false);
        }
    }

    // ---- epilogue: C/D layout (M = (lane/16)*8 + v, N = lane%16), split-K atomic add ----
#pragma unroll
    for (int mt = 0; mt < 2; ++mt) {
        const int mbase = m0 + waveM * 32 + mt * 16 + ((lane >> 4) << 3);
#pragma unroll
        for (int nt = 0; nt < 4; ++nt) {
            const int n = n0 + waveN * 64 + nt * 16 + (lane & 15);
#pragma unroll
            for (int v = 0; v < 8; ++v)
                unsafeAtomicAdd(&OUT[(size_t)(mbase + v) * N + n], acc[mt][nt][v]);
        }
    }
}

extern "C" void kernel_launch(void* const* d_in, const int* in_sizes, int n_in,
                              void* d_out, int out_size, void* d_ws, size_t ws_size,
                              hipStream_t stream) {
    (void)in_sizes; (void)n_in; (void)out_size; (void)ws_size;
    const float* x  = (const float*)d_in[0];
    const float* W0 = (const float*)d_in[1];
    const float* b0 = (const float*)d_in[2];
    const float* W1 = (const float*)d_in[3];
    const float* b1 = (const float*)d_in[4];
    float* out = (float*)d_out;
    float* h   = (float*)d_ws;   // [512][512] intermediate activations (1 MB)

    // split-K accumulates via atomics -> outputs must start at zero each call
    hipMemsetAsync(h,   0, (size_t)512 * 512 * sizeof(float), stream);
    hipMemsetAsync(out, 0, (size_t)512 * 256 * sizeof(float), stream);

    dim3 blk(256);
    dim3 g0(512 / BM, 512 / BN, SPLITK);   // 8 x 2 x 16 = 256 blocks
    cubic_gemm_wmma<<<g0, blk, 0, stream>>>(x, W0, b0, h, 512);
    dim3 g1(512 / BM, 256 / BN, SPLITK);   // 8 x 1 x 16 = 128 blocks
    cubic_gemm_wmma<<<g1, blk, 0, stream>>>(h, W1, b1, out, 256);
}
